// MemAttention_32693291057190
// MI455X (gfx1250) — matrix operand order repaired and checked by reading, not compiled, not run on hardware
//
#include <hip/hip_runtime.h>

typedef __attribute__((ext_vector_type(16))) _Float16 v16h;
typedef __attribute__((ext_vector_type(8)))  _Float16 v8h;
typedef __attribute__((ext_vector_type(8)))  float    v8f;

#define B_      2
#define H_      16
#define L_      1024
#define E_      64
#define KV_     2048
#define MEMTOT_ 1024          // mem_len + lmem_len
#define SCALE_  0.125f        // 1/sqrt(64)
#define SCALE_LOG2E_ 0.1803368801111244f  // 0.125 * log2(e): WMMA outputs in base-2
#define NEG_INF_ (-1.0e30f)

__device__ __forceinline__ float shfl32(float v, int srcLane) {
  return __builtin_bit_cast(
      float, __builtin_amdgcn_ds_bpermute(srcLane << 2, __builtin_bit_cast(int, v)));
}

__device__ __forceinline__ v8f wmma_f16(v16h a, v16h b, v8f c) {
  return __builtin_amdgcn_wmma_f32_16x16x32_f16(false, a, false, b, (short)0, c,
                                                false, false);
}

__device__ __forceinline__ v8f zero8() {
  v8f z = {0.f, 0.f, 0.f, 0.f, 0.f, 0.f, 0.f, 0.f};
  return z;
}

__device__ __forceinline__ float fast_exp2(float x) {
#if __has_builtin(__builtin_amdgcn_exp2f)
  return __builtin_amdgcn_exp2f(x);        // v_exp_f32 (native 2^x)
#else
  return __expf(x * 0.6931471805599453f);
#endif
}

// ============================================================================
// Pre-conversion kernels (one-shot, memory bound, L2-resident data)
// ============================================================================
__global__ void __launch_bounds__(256)
cvt_to_f16(const float* __restrict__ src, _Float16* __restrict__ dst,
           int nelem, float mul) {
  const int i = blockIdx.x * 256 + threadIdx.x;
  if (i < nelem) dst[i] = (_Float16)(src[i] * mul);
}

// v[b,h,t,e] f32 -> vt[b,h,e,t] f16  (coalesced writes along t)
__global__ void __launch_bounds__(256)
transpose_v_f16(const float* __restrict__ v, _Float16* __restrict__ vt) {
  const int i  = blockIdx.x * 256 + threadIdx.x;  // grid sized exactly
  const int t  = i & (KV_ - 1);
  const int e  = (i >> 11) & (E_ - 1);
  const int bh = i >> 17;
  vt[i] = (_Float16)v[((size_t)bh * KV_ + t) * E_ + e];
}

// ============================================================================
// Main flash-attention kernel, f16 operands from workspace.
// One wave32 per 16-row Q tile; fixed-shift base-2 softmax; rowsum via
// ones-WMMA; rolling pos-band tile; split main/tail loops (tail = 1 chunk).
// ============================================================================
__global__ void __launch_bounds__(256)
memattn_fa16(const _Float16* __restrict__ qh, const _Float16* __restrict__ kh,
             const _Float16* __restrict__ ph, const _Float16* __restrict__ vth,
             float* __restrict__ out)
{
  __shared__ _Float16 ldsP[8][16 * 56];   // probs staging (stride 56 halves)
  __shared__ float    ldsB[8][16 * 49];   // pos-bias band (16 rows x 48 cols)

  const int lane   = threadIdx.x & 31;
  const int wid    = threadIdx.x >> 5;
  const int tile   = blockIdx.x * 8 + wid;     // 0..2047
  const int r0     = (tile & 63) << 4;
  const int h      = (tile >> 6) & (H_ - 1);
  const int b      = tile >> 10;
  const int laneHi = lane >> 4;
  const int n      = lane & 15;

  const _Float16* qb = qh  + (size_t)(b * H_ + h) * L_  * E_;
  const _Float16* kb = kh  + (size_t)(b * H_ + h) * KV_ * E_;
  const _Float16* vt = vth + (size_t)(b * H_ + h) * E_  * KV_;
  const _Float16* pb = ph  + (size_t)h * KV_ * E_;

  // ---- resident Q A-fragments (0.125*log2e pre-folded in cvt kernel)
  v16h qa[2];
  {
    const _Float16* qrow = qb + (size_t)(r0 + n) * E_;
    #pragma unroll
    for (int ks = 0; ks < 2; ++ks) {
      const v8h lo = *(const v8h*)(qrow + 16 * laneHi + 32 * ks);
      const v8h hi = *(const v8h*)(qrow + 8 + 16 * laneHi + 32 * ks);
      qa[ks] = __builtin_shufflevector(lo, hi, 0, 1, 2, 3, 4, 5, 6, 7,
                                       8, 9, 10, 11, 12, 13, 14, 15);
    }
  }

  v16h ones;
  #pragma unroll
  for (int i = 0; i < 16; ++i) ones[i] = (_Float16)1.0f;

  v8f acc[4];
  #pragma unroll
  for (int et = 0; et < 4; ++et) acc[et] = zero8();
  v8f sumAcc = zero8();

  float* bb = &ldsB[wid][0];

  // --- helpers -------------------------------------------------------------
  auto posTile = [&](int c0) -> v8f {            // unclamped (main loop safe)
    const _Float16* prow = pb + (size_t)(c0 + n) * E_;
    const v16h f0 = *(const v16h*)(prow + 16 * laneHi);
    const v16h f1 = *(const v16h*)(prow + 32 + 16 * laneHi);
    v8f cz = zero8();
    cz = wmma_f16(qa[0], f0, cz);
    cz = wmma_f16(qa[1], f1, cz);
    return cz;
  };
  auto posTileClamped = [&](int c0) -> v8f {     // tail: c may exceed KV
    const int c = c0 + n;
    const bool ok = (c < KV_);
    const _Float16* prow = pb + (size_t)(ok ? c : 0) * E_;
    v16h f0 = *(const v16h*)(prow + 16 * laneHi);
    v16h f1 = *(const v16h*)(prow + 32 + 16 * laneHi);
    if (!ok) {
      #pragma unroll
      for (int i = 0; i < 16; ++i) { f0[i] = (_Float16)0.0f; f1[i] = (_Float16)0.0f; }
    }
    v8f cz = zero8();
    cz = wmma_f16(qa[0], f0, cz);
    cz = wmma_f16(qa[1], f1, cz);
    return cz;
  };
  auto storeTile = [&](v8f tv, int jt) {
    #pragma unroll
    for (int r = 0; r < 8; ++r)
      bb[(r + 8 * laneHi) * 49 + 16 * jt + n] = tv[r];
  };
  auto qkTiles = [&](int t0, v8f& s0, v8f& s1) {
    #pragma unroll
    for (int j = 0; j < 2; ++j) {
      const _Float16* krow = kb + (size_t)(t0 + 16 * j + n) * E_;
      const v16h f0 = *(const v16h*)(krow + 16 * laneHi);
      const v16h f1 = *(const v16h*)(krow + 32 + 16 * laneHi);
      v8f cz = zero8();
      cz = wmma_f16(qa[0], f0, cz);
      cz = wmma_f16(qa[1], f1, cz);
      if (j == 0) s0 = cz; else s1 = cz;
    }
  };
  auto finishChunk = [&](int t0, v8f s0, v8f s1, bool masked) {
    // bias add (+ tail causal mask), base-2 exp
    #pragma unroll
    for (int r = 0; r < 8; ++r) {
      const int m = r + 8 * laneHi;
      float a0 = s0[r] + bb[m * 49 + (15 + n - m)];
      float a1 = s1[r] + bb[m * 49 + (31 + n - m)];
      if (masked) {
        if (t0 + n      > r0 + m + MEMTOT_) a0 = NEG_INF_;
        if (t0 + 16 + n > r0 + m + MEMTOT_) a1 = NEG_INF_;
      }
      s0[r] = fast_exp2(a0);
      s1[r] = fast_exp2(a1);
    }
    // probs: C-layout f32 -> LDS f16 -> A-layout fragment
    _Float16* pl = &ldsP[wid][0];
    #pragma unroll
    for (int r = 0; r < 8; ++r) {
      const int m = r + 8 * laneHi;
      pl[m * 56 + n]      = (_Float16)s0[r];
      pl[m * 56 + 16 + n] = (_Float16)s1[r];
    }
    v16h pa;
    {
      const v8h lo = *(const v8h*)(pl + n * 56 + 16 * laneHi);
      const v8h hi = *(const v8h*)(pl + n * 56 + 8 + 16 * laneHi);
      pa = __builtin_shufflevector(lo, hi, 0, 1, 2, 3, 4, 5, 6, 7,
                                   8, 9, 10, 11, 12, 13, 14, 15);
    }
    // O += P.V (4 E-tiles) + rowsum WMMA
    #pragma unroll
    for (int et = 0; et < 4; ++et) {
      const v16h fv = *(const v16h*)(vt + (size_t)(16 * et + n) * KV_ +
                                     t0 + 16 * laneHi);
      acc[et] = wmma_f16(pa, fv, acc[et]);
    }
    sumAcc = wmma_f16(pa, ones, sumAcc);
  };
  // -------------------------------------------------------------------------

  int tEnd = r0 + 16 + MEMTOT_;
  if (tEnd > KV_) tEnd = KV_;
  const int tSplit = (r0 + MEMTOT_) & ~31;   // exactly one masked chunk after

  // rolling pos tile: tile[0] of chunk i+1 == tile[2] of chunk i
  v8f carry = posTile(0 - r0 + (MEMTOT_ - 16));

  for (int t0 = 0; t0 < tSplit; t0 += 32) {
    __builtin_prefetch(kb + (size_t)(t0 + 32 + n) * E_, 0, 0);
    __builtin_prefetch(kb + (size_t)(t0 + 48 + n) * E_, 0, 0);

    v8f s0, s1;
    qkTiles(t0, s0, s1);

    const int c_lo = t0 - r0 + (MEMTOT_ - 16);
    storeTile(carry, 0);
    const v8f p1 = posTile(c_lo + 16);
    storeTile(p1, 1);
    const v8f p2 = posTile(c_lo + 32);
    storeTile(p2, 2);
    carry = p2;

    finishChunk(t0, s0, s1, false);
  }

  for (int t0 = tSplit; t0 < tEnd; t0 += 32) {   // exactly 1 iteration
    v8f s0, s1;
    qkTiles(t0, s0, s1);
    const int c_lo = t0 - r0 + (MEMTOT_ - 16);
    storeTile(posTileClamped(c_lo), 0);
    storeTile(posTileClamped(c_lo + 16), 1);
    storeTile(posTileClamped(c_lo + 32), 2);
    finishChunk(t0, s0, s1, true);
  }

  // ===== epilogue: normalize, write out[b, r, h*64 + e] =====
  #pragma unroll
  for (int r = 0; r < 8; ++r) {
    const int   m   = r + 8 * laneHi;
    const float inv = 1.0f / sumAcc[r];
    float* orow = out + ((size_t)(b * L_ + (r0 + m)) * H_ + h) * E_;
    #pragma unroll
    for (int et = 0; et < 4; ++et) orow[16 * et + n] = acc[et][r] * inv;
  }
}

// ============================================================================
// Fallback: fully self-contained f32-input kernel (used if ws too small)
// ============================================================================
__global__ void __launch_bounds__(256)
memattn_fa(const float* __restrict__ q, const float* __restrict__ k,
           const float* __restrict__ v, const float* __restrict__ pos,
           float* __restrict__ out)
{
  __shared__ _Float16 ldsP[8][16 * 36];

  const int lane   = threadIdx.x & 31;
  const int wid    = threadIdx.x >> 5;
  const int tile   = blockIdx.x * 8 + wid;
  const int r0     = (tile & 63) << 4;
  const int h      = (tile >> 6) & (H_ - 1);
  const int b      = tile >> 10;
  const int laneHi = lane >> 4;
  const int n      = lane & 15;

  const float* qb = q   + (size_t)(b * H_ + h) * L_  * E_;
  const float* kb = k   + (size_t)(b * H_ + h) * KV_ * E_;
  const float* vb = v   + (size_t)(b * H_ + h) * KV_ * E_;
  const float* pp = pos + (size_t)h * KV_ * E_;

  v16h qa[2];
  {
    const float* qrow = qb + (size_t)(r0 + n) * E_;
    #pragma unroll
    for (int ks = 0; ks < 2; ++ks) {
      #pragma unroll
      for (int i = 0; i < 16; ++i) {
        const int e = i + 16 * laneHi + 32 * ks;
        qa[ks][i] = (_Float16)(qrow[e] * SCALE_);
      }
    }
  }

  v8f acc[4];
  float rowM[8], rowS[8];
  #pragma unroll
  for (int et = 0; et < 4; ++et) acc[et] = zero8();
  #pragma unroll
  for (int r = 0; r < 8; ++r) { rowM[r] = NEG_INF_; rowS[r] = 0.0f; }

  int tEnd = r0 + 16 + MEMTOT_;
  if (tEnd > KV_) tEnd = KV_;

  for (int t0 = 0; t0 < tEnd; t0 += 32) {
    v8f s[2];
    #pragma unroll
    for (int j = 0; j < 2; ++j) {
      int t = t0 + 16 * j + n;
      if (t >= KV_) t = KV_ - 1;
      const float* krow = kb + (size_t)t * E_ + 16 * laneHi;
      v16h fb0, fb1;
      #pragma unroll
      for (int i = 0; i < 16; ++i) fb0[i] = (_Float16)krow[i];
      #pragma unroll
      for (int i = 0; i < 16; ++i) fb1[i] = (_Float16)krow[32 + i];
      v8f cz = zero8();
      cz = wmma_f16(qa[0], fb0, cz);
      cz = wmma_f16(qa[1], fb1, cz);
      s[j] = cz;
    }

    const int c_lo = t0 - r0 + (MEMTOT_ - 16);
    v8f pband[3];
    #pragma unroll
    for (int jt = 0; jt < 3; ++jt) {
      const int c  = c_lo + 16 * jt + n;
      const bool ok = (c < KV_);
      const float* prow = pp + (size_t)(ok ? c : 0) * E_ + 16 * laneHi;
      v16h fb0, fb1;
      #pragma unroll
      for (int i = 0; i < 16; ++i) fb0[i] = ok ? (_Float16)prow[i] : (_Float16)0.0f;
      #pragma unroll
      for (int i = 0; i < 16; ++i) fb1[i] = ok ? (_Float16)prow[32 + i] : (_Float16)0.0f;
      v8f cz = zero8();
      cz = wmma_f16(qa[0], fb0, cz);
      cz = wmma_f16(qa[1], fb1, cz);
      pband[jt] = cz;
    }

    #pragma unroll
    for (int r = 0; r < 8; ++r) {
      const int m = r + 8 * laneHi;
      #pragma unroll
      for (int j = 0; j < 2; ++j) {
        const int ccoff   = 15 + n + 16 * j - m;
        const int srcLane = (ccoff & 15) | (laneHi << 4);
        const float p0 = shfl32(pband[0][r], srcLane);
        const float p1 = shfl32(pband[1][r], srcLane);
        const float p2 = shfl32(pband[2][r], srcLane);
        const int  sel = ccoff >> 4;
        const float pv = (sel == 0) ? p0 : ((sel == 1) ? p1 : p2);
        const int  tAbs  = t0 + 16 * j + n;
        const bool valid = (tAbs <= r0 + m + MEMTOT_) && (tAbs < KV_);
        s[j][r] = valid ? (s[j][r] + pv) : NEG_INF_;
      }
    }

    #pragma unroll
    for (int r = 0; r < 8; ++r) {
      float mx = fmaxf(s[0][r], s[1][r]);
      #pragma unroll
      for (int d = 1; d < 16; d <<= 1) mx = fmaxf(mx, shfl32(mx, lane ^ d));
      const float newM  = fmaxf(rowM[r], mx);
      const float alpha = __expf(rowM[r] - newM);
      rowM[r] = newM;
      const float e0 = __expf(s[0][r] - newM);
      const float e1 = __expf(s[1][r] - newM);
      s[0][r] = e0; s[1][r] = e1;
      float rs = e0 + e1;
      #pragma unroll
      for (int d = 1; d < 16; d <<= 1) rs += shfl32(rs, lane ^ d);
      rowS[r] = rowS[r] * alpha + rs;
      #pragma unroll
      for (int et = 0; et < 4; ++et) acc[et][r] *= alpha;
    }

    _Float16* pl = &ldsP[wid][0];
    #pragma unroll
    for (int r = 0; r < 8; ++r) {
      const int m = r + 8 * laneHi;
      pl[m * 36 + n]      = (_Float16)s[0][r];
      pl[m * 36 + 16 + n] = (_Float16)s[1][r];
    }
    v16h pa;
    #pragma unroll
    for (int i = 0; i < 16; ++i) {
      const int kk = i + 16 * laneHi;
      pa[i] = pl[n * 36 + kk];
    }

    #pragma unroll
    for (int et = 0; et < 4; ++et) {
      v16h fv;
      #pragma unroll
      for (int i = 0; i < 16; ++i) {
        int t = t0 + i + 16 * laneHi;
        if (t >= KV_) t = KV_ - 1;
        fv[i] = (_Float16)vb[(size_t)t * E_ + 16 * et + n];
      }
      acc[et] = wmma_f16(pa, fv, acc[et]);
    }
  }

  #pragma unroll
  for (int r = 0; r < 8; ++r) {
    const int   m   = r + 8 * laneHi;
    const float inv = 1.0f / rowS[r];
    float* orow = out + ((size_t)(b * L_ + (r0 + m)) * H_ + h) * E_;
    #pragma unroll
    for (int et = 0; et < 4; ++et) orow[16 * et + n] = acc[et][r] * inv;
  }
}

// ============================================================================
extern "C" void kernel_launch(void* const* d_in, const int* in_sizes, int n_in,
                              void* d_out, int out_size, void* d_ws, size_t ws_size,
                              hipStream_t stream) {
  (void)in_sizes; (void)n_in; (void)out_size;
  const float* q   = (const float*)d_in[0];
  const float* k   = (const float*)d_in[1];
  const float* v   = (const float*)d_in[2];
  const float* pos = (const float*)d_in[4];
  float* out = (float*)d_out;

  const int nQ = B_ * H_ * L_  * E_;   // 2,097,152
  const int nK = B_ * H_ * KV_ * E_;   // 4,194,304
  const int nP = H_ * KV_ * E_;        // 2,097,152
  const size_t need = (size_t)(nQ + nK + nP + nK) * sizeof(_Float16); // 24 MB

  const dim3 blk(256);
  if (ws_size >= need) {
    char* w = (char*)d_ws;
    _Float16* qh = (_Float16*)w;
    _Float16* kh = (_Float16*)(w + (size_t)nQ * 2);
    _Float16* ph = (_Float16*)(w + (size_t)(nQ + nK) * 2);
    _Float16* vt = (_Float16*)(w + (size_t)(nQ + nK + nP) * 2);

    cvt_to_f16<<<dim3((nQ + 255) / 256), blk, 0, stream>>>(q, qh, nQ, SCALE_LOG2E_);
    cvt_to_f16<<<dim3((nK + 255) / 256), blk, 0, stream>>>(k, kh, nK, 1.0f);
    cvt_to_f16<<<dim3((nP + 255) / 256), blk, 0, stream>>>(pos, ph, nP, 1.0f);
    transpose_v_f16<<<dim3(nK / 256), blk, 0, stream>>>(v, vt);

    memattn_fa16<<<dim3(B_ * H_ * (L_ / 16) / 8), blk, 0, stream>>>(qh, kh, ph, vt, out);
  } else {
    memattn_fa<<<dim3(B_ * H_ * (L_ / 16) / 8), blk, 0, stream>>>(q, k, v, pos, out);
  }
}
